// SA_53060025974801
// MI455X (gfx1250) — matrix operand order repaired and checked by reading, not compiled, not run on hardware
//
#include <hip/hip_runtime.h>

typedef __attribute__((ext_vector_type(16))) _Float16 v16h;
typedef __attribute__((ext_vector_type(8)))  float    v8f;

#define B_     4
#define N_     8192
#define S_     2048
#define G_     32
#define CFEA_  13
#define M_     (B_*S_*G_)      // 262144
#define R2_    (0.12f*0.12f)
#define EPS_   1e-5f

// ---------------------------------------------------------------- FPS
// One block per batch. 1024 threads, 8 points/thread held in registers.
__global__ __launch_bounds__(1024) void fps_kernel(const float* __restrict__ xyz,
                                                   float* __restrict__ newco,
                                                   float* __restrict__ distout,
                                                   float* __restrict__ sampout) {
    const int b = blockIdx.x, tid = threadIdx.x;
    const float* xb = xyz + (size_t)b * N_ * 3;
    __shared__ float red_v[1024];
    __shared__ int   red_i[1024];
    __shared__ float cent[3];

    float px[8], py[8], pz[8], dist[8];
#pragma unroll
    for (int p = 0; p < 8; ++p) {
        int j = tid * 8 + p;
        px[p] = xb[j*3+0]; py[p] = xb[j*3+1]; pz[p] = xb[j*3+2];
        dist[p] = 1e10f;
    }
    int far = 0;
    for (int s = 0; s < S_; ++s) {
        if (tid == 0) {
            float cx = xb[far*3+0], cy = xb[far*3+1], cz = xb[far*3+2];
            cent[0]=cx; cent[1]=cy; cent[2]=cz;
            size_t o = (size_t)b * S_ + s;
            sampout[o] = (float)far;
            newco[o*3+0]=cx; newco[o*3+1]=cy; newco[o*3+2]=cz;
        }
        __syncthreads();
        float cx = cent[0], cy = cent[1], cz = cent[2];
        float bestv = -1.0f; int besti = 0;
#pragma unroll
        for (int p = 0; p < 8; ++p) {
            float dx = px[p]-cx, dy = py[p]-cy, dz = pz[p]-cz;
            float d = dx*dx + dy*dy + dz*dz;
            dist[p] = fminf(dist[p], d);
            if (dist[p] > bestv) { bestv = dist[p]; besti = tid*8+p; }
        }
        red_v[tid] = bestv; red_i[tid] = besti;
        __syncthreads();
        for (int st = 512; st > 0; st >>= 1) {
            if (tid < st) {
                float v2 = red_v[tid+st]; int i2 = red_i[tid+st];
                if (v2 > red_v[tid] || (v2 == red_v[tid] && i2 < red_i[tid])) {
                    red_v[tid] = v2; red_i[tid] = i2;
                }
            }
            __syncthreads();
        }
        far = red_i[0];
    }
#pragma unroll
    for (int p = 0; p < 8; ++p) distout[(size_t)b*N_ + tid*8+p] = dist[p];
}

// ------------------------------------------- ball query + build x0 (f16, K padded to 32)
__device__ __forceinline__ void write_row(_Float16* dst, const float* __restrict__ xb,
                                          const float* __restrict__ fb, int j,
                                          float cx, float cy, float cz) {
    union { _Float16 h[32]; uint4 q[4]; } r;
    const float* p = xb + (size_t)j*3;
    const float* f = fb + (size_t)j*CFEA_;
    r.h[0] = (_Float16)(p[0]-cx); r.h[1] = (_Float16)(p[1]-cy); r.h[2] = (_Float16)(p[2]-cz);
#pragma unroll
    for (int k = 0; k < CFEA_; ++k) r.h[3+k] = (_Float16)f[k];
#pragma unroll
    for (int k = 16; k < 32; ++k) r.h[k] = (_Float16)0.0f;
    uint4* d = (uint4*)dst;
    d[0]=r.q[0]; d[1]=r.q[1]; d[2]=r.q[2]; d[3]=r.q[3];
}

__global__ __launch_bounds__(256) void ballq_kernel(const float* __restrict__ xyz,
                                                    const float* __restrict__ fea,
                                                    const float* __restrict__ newco,
                                                    _Float16* __restrict__ x0) {
    const int lane = threadIdx.x & 31;
    const int wid  = blockIdx.x * 8 + (threadIdx.x >> 5);
    const int b = wid / S_;
    const float* xb = xyz + (size_t)b * N_ * 3;
    const float* fb = fea + (size_t)b * N_ * CFEA_;
    const size_t co = (size_t)wid;
    const float cx = newco[co*3+0], cy = newco[co*3+1], cz = newco[co*3+2];
    _Float16* rowbase = x0 + co * G_ * 32;

    int count = 0, j0 = -1;
    for (int base = 0; base < N_ && count < G_; base += 32) {
        int j = base + lane;
        float dx = xb[j*3+0]-cx, dy = xb[j*3+1]-cy, dz = xb[j*3+2]-cz;
        float d2 = dx*dx + dy*dy + dz*dz;
        bool pred = d2 <= R2_;
        unsigned mask = __builtin_amdgcn_ballot_w32(pred);
        if (j0 < 0 && mask) j0 = base + __builtin_ctz(mask);
        if (pred) {
            int pos = count + __popc(mask & ((1u << lane) - 1u));
            if (pos < G_) write_row(rowbase + (size_t)pos * 32, xb, fb, j, cx, cy, cz);
        }
        count += __popc(mask);
    }
    int cnt = count < G_ ? count : G_;
    if (j0 >= 0 && lane >= cnt) // duplicate first neighbor into remaining slots
        write_row(rowbase + (size_t)lane * 32, xb, fb, j0, cx, cy, cz);
}

// ---------------------------------------- pack weights into WMMA B-matrix lane layout
// dst[((c*KC+kc)*32 + L)*16 + e] = W[(c*16 + (L&15)) * Kin + kc*32 + 8*(L>>4) + (e&7) + 16*(e>>3)]
__global__ void pack_w(const float* __restrict__ W, _Float16* __restrict__ dst,
                       int Kin, int Nc, int KC) {
    int tid = blockIdx.x * blockDim.x + threadIdx.x;
    int total = Nc * KC * 32 * 16;
    if (tid >= total) return;
    int e  = tid & 15;
    int L  = (tid >> 4) & 31;
    int gk = tid >> 9;               // c*KC + kc
    int kc = gk % KC, c = gk / KC;
    int n = c * 16 + (L & 15);
    int k = kc * 32 + 8 * (L >> 4) + (e & 7) + 16 * (e >> 3);
    float v = (k < Kin) ? W[(size_t)n * Kin + k] : 0.0f;
    dst[tid] = (_Float16)v;
}

__global__ void zero_stats(float* __restrict__ st) {
    st[blockIdx.x * 256 + threadIdx.x] = 0.0f;
}

// ------------- GEMM (WMMA f16->f32) + bias + BN-stat atomics, fully unrolled via templates
template <int KPAD, int KC, int NOUT, int NC>
__global__ __launch_bounds__(256) void gemm_bn_stats(const _Float16* __restrict__ A,
                                                     const _Float16* __restrict__ Bp,
                                                     const float* __restrict__ bias,
                                                     _Float16* __restrict__ Y,
                                                     float* __restrict__ stats) {
    __shared__ float sst[256];                 // [0..NOUT) sums, [NOUT..2*NOUT) sumsq
    const int t = threadIdx.x;
    sst[t] = 0.0f;
    __syncthreads();

    const int lane = t & 31, wv = t >> 5;
    const int h = lane >> 4;
    const int base = (blockIdx.x * 8 + wv) * 16;
    const int row  = base + (lane & 15);

    union AV { v16h v; uint4 q[2]; };
    AV a[KC];
    const _Float16* asrc = A + (size_t)row * KPAD + 8 * h;   // per-lane A base
#pragma unroll
    for (int kc = 0; kc < KC; ++kc) {          // A fragment: two contiguous 16B chunks
        a[kc].q[0] = *(const uint4*)(asrc + kc * 32);
        a[kc].q[1] = *(const uint4*)(asrc + kc * 32 + 16);
    }
    const int n_lo = lane & 15;
    const _Float16* bbase = Bp + (size_t)lane * 16;          // per-lane B base
    _Float16* ybase = Y + (size_t)(base + 8 * h) * NOUT + n_lo;
#pragma unroll
    for (int c = 0; c < NC; ++c) {
        v8f acc = {};
#pragma unroll
        for (int kc = 0; kc < KC; ++kc) {
            AV bm;
            bm.q[0] = *(const uint4*)(bbase + (c * KC + kc) * 512);
            bm.q[1] = *(const uint4*)(bbase + (c * KC + kc) * 512 + 8);
            acc = __builtin_amdgcn_wmma_f32_16x16x32_f16(
                      false, a[kc].v, false, bm.v, (short)0, acc, false, false);
        }
        const float bn = bias[c * 16 + n_lo];
        float lsum = 0.0f, lsq = 0.0f;
#pragma unroll
        for (int r = 0; r < 8; ++r) {
            float v = acc[r] + bn;
            ybase[(size_t)r * NOUT + c * 16] = (_Float16)v;
            lsum += v; lsq += v * v;
        }
        atomicAdd(&sst[c * 16 + n_lo], lsum);
        atomicAdd(&sst[NOUT + c * 16 + n_lo], lsq);
    }
    __syncthreads();
    if (t < 2 * NOUT) atomicAdd(&stats[t], sst[t]);
}

__global__ void bn_finalize(const float* __restrict__ stats, const float* __restrict__ g,
                            const float* __restrict__ be, float* __restrict__ scsh,
                            int Nout, float invM) {
    int c = threadIdx.x;
    if (c < Nout) {
        float mu  = stats[c] * invM;
        float var = stats[Nout + c] * invM - mu * mu;
        float inv = rsqrtf(var + EPS_);
        float sc  = g[c] * inv;
        scsh[c]        = sc;
        scsh[Nout + c] = be[c] - mu * sc;
    }
}

__global__ void bn_relu_inplace(_Float16* __restrict__ Y, const float* __restrict__ scsh,
                                int mask, int Nout, size_t total) {
    size_t stride = (size_t)gridDim.x * blockDim.x;
    for (size_t i = (size_t)blockIdx.x * blockDim.x + threadIdx.x; i < total; i += stride) {
        int c = (int)(i & (size_t)mask);
        float v = (float)Y[i];
        v = fmaxf(v * scsh[c] + scsh[Nout + c], 0.0f);
        Y[i] = (_Float16)v;
    }
}

__global__ __launch_bounds__(128) void group_max(const _Float16* __restrict__ act,
                                                 float* __restrict__ out) {
    const int bs = blockIdx.x, c = threadIdx.x;
    float m = 0.0f;                                        // post-ReLU values are >= 0
#pragma unroll
    for (int g = 0; g < G_; ++g)
        m = fmaxf(m, (float)act[((size_t)bs * G_ + g) * 128 + c]);
    out[(size_t)bs * 128 + c] = m;
}

// ---------------------------------------------------------------- launcher
extern "C" void kernel_launch(void* const* d_in, const int* in_sizes, int n_in,
                              void* d_out, int out_size, void* d_ws, size_t ws_size,
                              hipStream_t stream) {
    const float* xyz = (const float*)d_in[0];
    const float* fea = (const float*)d_in[1];
    const float* W0 = (const float*)d_in[2],  *b0 = (const float*)d_in[3];
    const float* g0 = (const float*)d_in[4],  *be0 = (const float*)d_in[5];
    const float* W1 = (const float*)d_in[6],  *b1 = (const float*)d_in[7];
    const float* g1 = (const float*)d_in[8],  *be1 = (const float*)d_in[9];
    const float* W2 = (const float*)d_in[10], *b2 = (const float*)d_in[11];
    const float* g2 = (const float*)d_in[12], *be2 = (const float*)d_in[13];

    float* out    = (float*)d_out;
    float* newco  = out;                                  // [B,S,3]
    float* newfea = out + (size_t)B_ * S_ * 3;            // [B,S,128]
    float* dist   = newfea + (size_t)B_ * S_ * 128;       // [B,N]
    float* samp   = dist + (size_t)B_ * N_;               // [B,S] (int cast to float)

    char* ws = (char*)d_ws;
    _Float16* x0   = (_Float16*)ws;                                   // M*32 f16 = 16 MB
    _Float16* bufA = (_Float16*)(ws + (size_t)M_ * 32 * 2);           // M*128 f16 = 64 MB
    _Float16* bufB = bufA + (size_t)M_ * 128;                         // M*128 f16 = 64 MB
    _Float16* pw0  = bufB + (size_t)M_ * 128;                         // packed weights
    _Float16* pw1  = pw0 + 4 * 1 * 512;
    _Float16* pw2  = pw1 + 4 * 2 * 512;
    float* stats   = (float*)(pw2 + 8 * 2 * 512);                     // 3 x 256 f32
    float* scsh    = stats + 768;                                     // 3 x 256 f32

    const float invM = 1.0f / (float)M_;

    fps_kernel<<<B_, 1024, 0, stream>>>(xyz, newco, dist, samp);
    ballq_kernel<<<B_ * S_ / 8, 256, 0, stream>>>(xyz, fea, newco, x0);

    pack_w<<<(4*1*512 + 255) / 256, 256, 0, stream>>>(W0, pw0, 16, 4, 1);
    pack_w<<<(4*2*512 + 255) / 256, 256, 0, stream>>>(W1, pw1, 64, 4, 2);
    pack_w<<<(8*2*512 + 255) / 256, 256, 0, stream>>>(W2, pw2, 64, 8, 2);
    zero_stats<<<3, 256, 0, stream>>>(stats);

    // Layer 0: x0 [M,32 padded from 16] -> bufA [M,64]
    gemm_bn_stats<32, 1, 64, 4><<<M_ / 128, 256, 0, stream>>>(x0, pw0, b0, bufA, stats);
    bn_finalize<<<1, 128, 0, stream>>>(stats, g0, be0, scsh, 64, invM);
    bn_relu_inplace<<<2048, 256, 0, stream>>>(bufA, scsh, 63, 64, (size_t)M_ * 64);

    // Layer 1: bufA [M,64] -> bufB [M,64]
    gemm_bn_stats<64, 2, 64, 4><<<M_ / 128, 256, 0, stream>>>(bufA, pw1, b1, bufB, stats + 256);
    bn_finalize<<<1, 128, 0, stream>>>(stats + 256, g1, be1, scsh + 256, 64, invM);
    bn_relu_inplace<<<2048, 256, 0, stream>>>(bufB, scsh + 256, 63, 64, (size_t)M_ * 64);

    // Layer 2: bufB [M,64] -> bufA [M,128]
    gemm_bn_stats<64, 2, 128, 8><<<M_ / 128, 256, 0, stream>>>(bufB, pw2, b2, bufA, stats + 512);
    bn_finalize<<<1, 128, 0, stream>>>(stats + 512, g2, be2, scsh + 512, 128, invM);
    bn_relu_inplace<<<2048, 256, 0, stream>>>(bufA, scsh + 512, 127, 128, (size_t)M_ * 128);

    group_max<<<B_ * S_, 128, 0, stream>>>(bufA, newfea);
}
